// MACELayer_42614665511391
// MI455X (gfx1250) — matrix operand order repaired and checked by reading, not compiled, not run on hardware
//
#include <hip/hip_runtime.h>
#include <hip/hip_bf16.h>
#include <math.h>

#define N_NODES 32768
#define N_EDGES 524288
#define F       128
#define NB      8
#define NP      5
#define ZSP     10
#define RW      16
#define NF      (N_NODES * F)

typedef __attribute__((ext_vector_type(16))) __bf16 v16bf;
typedef __attribute__((ext_vector_type(8)))  __bf16 v8bf;
typedef __attribute__((ext_vector_type(8)))  float  v8f;
typedef __attribute__((ext_vector_type(4)))  float  v4f;
typedef __attribute__((ext_vector_type(4)))  int    v4i;

#define INV128 0.08838834764831845f   // 1/sqrt(128)
#define INV8S  0.35355339059327373f   // 1/sqrt(8)
#define INV64  0.125f                 // 1/sqrt(64)
#define INV3S  0.5773502691896258f    // 1/sqrt(3)
#define INV2S  0.7071067811865476f    // 1/sqrt(2)
#define INVZ   0.31622776601683794f   // 1/sqrt(10)
#define INVFZ  0.027950849718747374f  // 1/sqrt(128*10)
#define INVR   0.25f                  // 1/sqrt(16)
#define EPSN   0.125f

// bf16 weight arena layout inside d_ws (element offsets)
#define OFF_UP0  0         // up0^T   [128][128]
#define OFF_UP1  16384     // up1^T   [128][128]
#define OFF_D0   32768     // down0^T [128][128]
#define OFF_D1   49152     // down1^T [128][128]
#define OFF_POST 65536     // post^T  [128][128]
#define OFF_R1   81920     // read1^T [16][128]
#define OFF_W1   83968     // mlp1^T  [64][32] (K padded 8->32)
#define OFF_W2   86016     // mlp2^T  [64][64]
#define OFF_W3   90112     // mlp3^T  [64][64]
#define OFF_W4   94208     // mlp4^T  [640][64]
#define WBUF_ELEMS 135168

__device__ __forceinline__ float silu_f(float x) { return x / (1.0f + __expf(-x)); }

// A-matrix 16x32 bf16 fragment from row-major [rows][ld] LDS (used for W^T).
// lane holds row M=lane&15; two contiguous 8-element runs -> 2x ds_load_b128.
__device__ __forceinline__ v16bf load_a_frag(const __bf16* base, int ld, int k0, int lane) {
  const int row = lane & 15;
  const int kg  = (lane >> 4) << 3;
  const __bf16* p = base + row * ld + k0 + kg;
  v8bf lo = *(const v8bf*)p;
  v8bf hi = *(const v8bf*)(p + 16);
  return __builtin_shufflevector(lo, hi, 0,1,2,3,4,5,6,7,8,9,10,11,12,13,14,15);
}

// B-matrix 32x16 fragment where B[k][n] = act[n][k], act row-major [16][ld].
// lane holds col n=lane&15; two contiguous 8-element runs -> 2x ds_load_b128.
__device__ __forceinline__ v16bf load_b_act(const __bf16* act, int ld, int k0, int lane) {
  const int n  = lane & 15;
  const int kg = (lane >> 4) << 3;
  const __bf16* p = act + n * ld + k0 + kg;
  v8bf lo = *(const v8bf*)p;
  v8bf hi = *(const v8bf*)(p + 16);
  return __builtin_shufflevector(lo, hi, 0,1,2,3,4,5,6,7,8,9,10,11,12,13,14,15);
}

__device__ __forceinline__ v8f wmma_bf16(v16bf a, v16bf b, v8f c) {
  return __builtin_amdgcn_wmma_f32_16x16x32_bf16(false, a, false, b, (short)0, c, false, false);
}

// 16B-granular global(bf16) -> LDS copy; async-to-LDS path when toolchain has it.
#if __has_builtin(__builtin_amdgcn_global_load_async_to_lds_b128) && \
    __has_builtin(__builtin_amdgcn_s_wait_asynccnt)
#define HAVE_ASYNC_LDS 1
typedef __attribute__((address_space(1))) v4i* g_v4i;
typedef __attribute__((address_space(3))) v4i* l_v4i;
#endif

__device__ __forceinline__ void copy_lds_16B(__bf16* dst, const __bf16* __restrict__ src,
                                             int elems, int tid, int nth) {
#ifdef HAVE_ASYNC_LDS
  for (int i = tid * 8; i < elems; i += nth * 8) {
    v4i* s = (v4i*)(src + i);   // C-style: const-cast + reinterpret (generic AS)
    v4i* d = (v4i*)(dst + i);
    __builtin_amdgcn_global_load_async_to_lds_b128((g_v4i)s, (l_v4i)d, 0, 0);
  }
#else
  for (int i = tid * 8; i < elems; i += nth * 8)
    *(v8bf*)(dst + i) = *(const v8bf*)(src + i);
#endif
}
__device__ __forceinline__ void copy_lds_fence() {
#ifdef HAVE_ASYNC_LDS
  __builtin_amdgcn_s_wait_asynccnt(0);
#endif
}

// ---------------------------------------------------------------------------
// Kernel 0: one-shot weight prep: f32 [K][N] -> bf16 transposed [N][K] arena.
// ---------------------------------------------------------------------------
__device__ __forceinline__ void prep_T(__bf16* dst, const float* __restrict__ src,
                                       int K, int N, int tid, int nth) {
  for (int i = tid; i < K * N; i += nth) {
    int n = i / K, k = i - n * K;
    dst[i] = (__bf16)src[k * N + n];
  }
}

__global__ __launch_bounds__(256) void k_prep_weights(
    const float* __restrict__ w_up0, const float* __restrict__ w_up1,
    const float* __restrict__ mw1, const float* __restrict__ mw2,
    const float* __restrict__ mw3, const float* __restrict__ mw4,
    const float* __restrict__ w_down0, const float* __restrict__ w_down1,
    const float* __restrict__ w_post, const float* __restrict__ w_read1,
    __bf16* __restrict__ wb)
{
  const int tid = blockIdx.x * 256 + threadIdx.x;
  const int nth = gridDim.x * 256;
  prep_T(wb + OFF_UP0,  w_up0,   F, F,  tid, nth);
  prep_T(wb + OFF_UP1,  w_up1,   F, F,  tid, nth);
  prep_T(wb + OFF_D0,   w_down0, F, F,  tid, nth);
  prep_T(wb + OFF_D1,   w_down1, F, F,  tid, nth);
  prep_T(wb + OFF_POST, w_post,  F, F,  tid, nth);
  prep_T(wb + OFF_R1,   w_read1, F, RW, tid, nth);
  // mlp_w1^T with K padded 8->32
  for (int i = tid; i < 64 * 32; i += nth) {
    int n = i >> 5, k = i & 31;
    wb[OFF_W1 + i] = (k < NB) ? (__bf16)mw1[k * 64 + n] : (__bf16)0.0f;
  }
  prep_T(wb + OFF_W2, mw2, 64, 64,  tid, nth);
  prep_T(wb + OFF_W3, mw3, 64, 64,  tid, nth);
  prep_T(wb + OFF_W4, mw4, 64, 640, tid, nth);
}

// ---------------------------------------------------------------------------
// Kernel 1: linear_up.  hs = ns @ w_up0 * inv ; hv[c] = nv[:,:,c] @ w_up1 * inv.
// 8 waves/block, 16 node-rows per wave; grid-stride over 128-row groups.
// ---------------------------------------------------------------------------
__global__ __launch_bounds__(256) void k_node_up(
    const float* __restrict__ node_scalars, const float* __restrict__ node_vectors,
    const __bf16* __restrict__ wb,
    float* __restrict__ hs, float* __restrict__ hv)
{
  __shared__ alignas(32) __bf16 s_w0T[F * F];
  __shared__ alignas(32) __bf16 s_w1T[F * F];
  __shared__ alignas(32) __bf16 s_in[8][16 * F];

  const int tid  = threadIdx.x;
  const int wv   = tid >> 5;
  const int lane = tid & 31;
  const int col  = lane & 15;
  const int mg   = (lane >> 4) << 3;

  copy_lds_16B(s_w0T, wb + OFF_UP0, F * F, tid, 256);
  copy_lds_16B(s_w1T, wb + OFF_UP1, F * F, tid, 256);
  copy_lds_fence();

  for (int g = blockIdx.x; g < N_NODES / 128; g += gridDim.x) {
    const int r0 = g * 128 + wv * 16;
    __syncthreads();
    for (int i = lane; i < 16 * (F / 8); i += 32) {
      int r = i >> 4, c8 = (i & 15) << 3;
      const float* src = node_scalars + (size_t)(r0 + r) * F + c8;
      v8bf v;
#pragma unroll
      for (int j = 0; j < 8; ++j) v[j] = (__bf16)src[j];
      *(v8bf*)(s_in[wv] + r * F + c8) = v;
    }
    __syncthreads();

#pragma unroll 1
    for (int t = 0; t < 8; ++t) {
      v8f acc = {};
#pragma unroll
      for (int ks = 0; ks < 4; ++ks)
        acc = wmma_bf16(load_a_frag(s_w0T + t * 16 * F, F, ks * 32, lane),
                        load_b_act(s_in[wv], F, ks * 32, lane), acc);
      float* dst = hs + (size_t)(r0 + col) * F + t * 16 + mg;
      v4f o0, o1;
#pragma unroll
      for (int r = 0; r < 4; ++r) { o0[r] = acc[r] * INV128; o1[r] = acc[4 + r] * INV128; }
      *(v4f*)dst = o0; *(v4f*)(dst + 4) = o1;
    }

    for (int c = 0; c < 3; ++c) {
      __syncthreads();
      for (int i = lane; i < 16 * (F / 8); i += 32) {
        int r = i >> 4, c8 = (i & 15) << 3;
        v8bf v;
#pragma unroll
        for (int j = 0; j < 8; ++j)
          v[j] = (__bf16)node_vectors[((size_t)(r0 + r) * F + c8 + j) * 3 + c];
        *(v8bf*)(s_in[wv] + r * F + c8) = v;
      }
      __syncthreads();
#pragma unroll 1
      for (int t = 0; t < 8; ++t) {
        v8f acc = {};
#pragma unroll
        for (int ks = 0; ks < 4; ++ks)
          acc = wmma_bf16(load_a_frag(s_w1T + t * 16 * F, F, ks * 32, lane),
                          load_b_act(s_in[wv], F, ks * 32, lane), acc);
        float* dst = hv + (size_t)c * NF + (size_t)(r0 + col) * F + t * 16 + mg;
        v4f o0, o1;
#pragma unroll
        for (int r = 0; r < 4; ++r) { o0[r] = acc[r] * INV128; o1[r] = acc[4 + r] * INV128; }
        *(v4f*)dst = o0; *(v4f*)(dst + 4) = o1;
      }
    }
  }
}

// ---------------------------------------------------------------------------
// Kernel 2: species-indexed skip linear (per-row weight -> VALU kernel).
// ---------------------------------------------------------------------------
__global__ __launch_bounds__(256) void k_self_conn(
    const float* __restrict__ node_scalars, const float* __restrict__ w_skip,
    const int* __restrict__ species, float* __restrict__ self_conn)
{
  __shared__ float xrow[2][F];
  const int sub = threadIdx.x >> 7;
  const int g   = threadIdx.x & 127;
  const int n   = blockIdx.x * 2 + sub;
  xrow[sub][g] = node_scalars[n * F + g];
  __syncthreads();
  const int z = species[n];
  const float* wz = w_skip + (size_t)z * F * F;
  float acc = 0.f;
#pragma unroll 4
  for (int f = 0; f < F; ++f) acc += xrow[sub][f] * wz[f * F + g];
  self_conn[n * F + g] = acc * INVFZ;
}

// ---------------------------------------------------------------------------
// Kernel 3: fused radial MLP (WMMA) + tensor product + scatter.
// 4 waves/block, 16 edges/wave; grid-stride over 64-edge groups.
// ---------------------------------------------------------------------------
__global__ __launch_bounds__(128) void k_edge(
    const float* __restrict__ vectors, const float* __restrict__ radial,
    const __bf16* __restrict__ wb,
    const int* __restrict__ senders, const int* __restrict__ receivers,
    const float* __restrict__ hs, const float* __restrict__ hv,
    float* __restrict__ agg_s, float* __restrict__ agg_v)
{
  __shared__ alignas(32) __bf16 s_w1T[64 * 32];
  __shared__ alignas(32) __bf16 s_w2T[64 * 64];
  __shared__ alignas(32) __bf16 s_w3T[64 * 64];
  __shared__ alignas(32) __bf16 s_w4T[640 * 64];
  __shared__ alignas(32) __bf16 s_a[4][16 * 64];
  __shared__ alignas(32) __bf16 s_b[4][16 * 64];
  __shared__ alignas(32) float  s_mix[4][NP][16 * 16];
  __shared__ float  s_y[4][16][3];
  __shared__ int    s_snd[4][16];
  __shared__ int    s_rcv[4][16];

  const int tid  = threadIdx.x;
  const int wv   = tid >> 5;
  const int lane = tid & 31;
  const int col  = lane & 15;
  const int mg   = (lane >> 4) << 3;

  copy_lds_16B(s_w1T, wb + OFF_W1, 64 * 32, tid, 128);
  copy_lds_16B(s_w2T, wb + OFF_W2, 64 * 64, tid, 128);
  copy_lds_16B(s_w3T, wb + OFF_W3, 64 * 64, tid, 128);
  copy_lds_16B(s_w4T, wb + OFF_W4, 640 * 64, tid, 128);
  copy_lds_fence();

  for (int g = blockIdx.x; g < N_EDGES / 64; g += gridDim.x) {
    const int e0 = g * 64 + wv * 16;
    __syncthreads();

    if (lane < 16) {
      int e = e0 + lane;
      s_snd[wv][lane] = senders[e];
      s_rcv[wv][lane] = receivers[e];
      float vx = vectors[e * 3 + 0], vy = vectors[e * 3 + 1], vz = vectors[e * 3 + 2];
      float rn = 1.0f / (sqrtf(vx * vx + vy * vy + vz * vz) + 1e-9f);
      s_y[wv][lane][0] = vx * rn; s_y[wv][lane][1] = vy * rn; s_y[wv][lane][2] = vz * rn;
    }
    for (int i = lane; i < 16 * 4; i += 32) {     // radial rows, K padded to 32
      int r = i >> 2, kc = i & 3;
      v8bf v;
#pragma unroll
      for (int j = 0; j < 8; ++j) v[j] = (kc == 0) ? (__bf16)radial[(e0 + r) * NB + j] : (__bf16)0.0f;
      *(v8bf*)(s_a[wv] + r * 32 + kc * 8) = v;
    }
    __syncthreads();

    // layer1: A=w1^T, B=radial act (ld 32), silu
#pragma unroll 1
    for (int t = 0; t < 4; ++t) {
      v8f acc = {};
      acc = wmma_bf16(load_a_frag(s_w1T + t * 16 * 32, 32, 0, lane),
                      load_b_act(s_a[wv], 32, 0, lane), acc);
      v8bf o;
#pragma unroll
      for (int r = 0; r < 8; ++r) o[r] = (__bf16)silu_f(acc[r] * INV8S);
      *(v8bf*)(s_b[wv] + col * 64 + t * 16 + mg) = o;
    }
    __syncthreads();
    // layer2: s_b -> s_a
#pragma unroll 1
    for (int t = 0; t < 4; ++t) {
      v8f acc = {};
#pragma unroll
      for (int ks = 0; ks < 2; ++ks)
        acc = wmma_bf16(load_a_frag(s_w2T + t * 16 * 64, 64, ks * 32, lane),
                        load_b_act(s_b[wv], 64, ks * 32, lane), acc);
      v8bf o;
#pragma unroll
      for (int r = 0; r < 8; ++r) o[r] = (__bf16)silu_f(acc[r] * INV64);
      *(v8bf*)(s_a[wv] + col * 64 + t * 16 + mg) = o;
    }
    __syncthreads();
    // layer3: s_a -> s_b
#pragma unroll 1
    for (int t = 0; t < 4; ++t) {
      v8f acc = {};
#pragma unroll
      for (int ks = 0; ks < 2; ++ks)
        acc = wmma_bf16(load_a_frag(s_w3T + t * 16 * 64, 64, ks * 32, lane),
                        load_b_act(s_a[wv], 64, ks * 32, lane), acc);
      v8bf o;
#pragma unroll
      for (int r = 0; r < 8; ++r) o[r] = (__bf16)silu_f(acc[r] * INV64);
      *(v8bf*)(s_b[wv] + col * 64 + t * 16 + mg) = o;
    }
    __syncthreads();

    // per channel-tile: layer4 (5 path tiles) then tensor product + scatter
#pragma unroll 1
    for (int t = 0; t < 8; ++t) {
#pragma unroll 1
      for (int p = 0; p < NP; ++p) {
        v8f acc = {};
#pragma unroll
        for (int ks = 0; ks < 2; ++ks)
          acc = wmma_bf16(load_a_frag(s_w4T + (p * F + t * 16) * 64, 64, ks * 32, lane),
                          load_b_act(s_b[wv], 64, ks * 32, lane), acc);
        float* mp = &s_mix[wv][p][col * 16 + mg];
        v4f o0, o1;
#pragma unroll
        for (int r = 0; r < 4; ++r) { o0[r] = acc[r] * INV64; o1[r] = acc[4 + r] * INV64; }
        *(v4f*)mp = o0; *(v4f*)(mp + 4) = o1;
      }
      __syncthreads();
#pragma unroll 1
      for (int i = 0; i < 8; ++i) {
        int pi = i * 32 + lane;
        int el = pi >> 4, fl = pi & 15;
        int f  = t * 16 + fl;
        int sn = s_snd[wv][el], rc = s_rcv[wv][el];
        float es  = hs[(size_t)sn * F + f];
        float ev0 = hv[0 * (size_t)NF + (size_t)sn * F + f];
        float ev1 = hv[1 * (size_t)NF + (size_t)sn * F + f];
        float ev2 = hv[2 * (size_t)NF + (size_t)sn * F + f];
        float y0 = s_y[wv][el][0], y1 = s_y[wv][el][1], y2 = s_y[wv][el][2];
        float m0c = s_mix[wv][0][el * 16 + fl];
        float m1c = s_mix[wv][1][el * 16 + fl];
        float m2c = s_mix[wv][2][el * 16 + fl];
        float m3c = s_mix[wv][3][el * 16 + fl];
        float m4c = s_mix[wv][4][el * 16 + fl];
        float dot = ev0 * y0 + ev1 * y1 + ev2 * y2;
        float m0  = m0c * es + m1c * dot * INV3S;
        float cx = ev1 * y2 - ev2 * y1;
        float cy = ev2 * y0 - ev0 * y2;
        float cz = ev0 * y1 - ev1 * y0;
        float b3 = m2c * es;
        atomicAdd(&agg_s[(size_t)rc * F + f], m0 * EPSN);
        atomicAdd(&agg_v[0 * (size_t)NF + (size_t)rc * F + f], (b3 * y0 + m3c * ev0 + m4c * cx * INV2S) * EPSN);
        atomicAdd(&agg_v[1 * (size_t)NF + (size_t)rc * F + f], (b3 * y1 + m3c * ev1 + m4c * cy * INV2S) * EPSN);
        atomicAdd(&agg_v[2 * (size_t)NF + (size_t)rc * F + f], (b3 * y2 + m3c * ev2 + m4c * cz * INV2S) * EPSN);
      }
      __syncthreads();
    }
  }
}

// ---------------------------------------------------------------------------
// Kernel 4: linear_down + symmetric contraction + post + skip + readout.
// 4 waves/block; all four weights persistent in LDS; grid-stride row groups.
// ---------------------------------------------------------------------------
__global__ __launch_bounds__(128) void k_node_down(
    const float* __restrict__ agg_s, const float* __restrict__ agg_v,
    const __bf16* __restrict__ wb,
    const float* __restrict__ w_sc, const float* __restrict__ w_read2,
    const float* __restrict__ self_conn, const int* __restrict__ species,
    float* __restrict__ out, float* __restrict__ feats_out)
{
  __shared__ alignas(32) __bf16 s_wd0[F * F];
  __shared__ alignas(32) __bf16 s_wd1[F * F];
  __shared__ alignas(32) __bf16 s_wp[F * F];
  __shared__ alignas(32) __bf16 s_wr1[RW * F];
  __shared__ float  s_wr2[RW];
  __shared__ alignas(32) __bf16 s_in[4][16 * F];
  __shared__ alignas(32) float  s_xs[4][16 * F];
  __shared__ alignas(32) float  s_vn[4][16 * F];

  const int tid  = threadIdx.x;
  const int wv   = tid >> 5;
  const int lane = tid & 31;
  const int col  = lane & 15;
  const int mg   = (lane >> 4) << 3;

  copy_lds_16B(s_wd0, wb + OFF_D0,   F * F,  tid, 128);
  copy_lds_16B(s_wd1, wb + OFF_D1,   F * F,  tid, 128);
  copy_lds_16B(s_wp,  wb + OFF_POST, F * F,  tid, 128);
  copy_lds_16B(s_wr1, wb + OFF_R1,   RW * F, tid, 128);
  if (tid < RW) s_wr2[tid] = w_read2[tid];
  copy_lds_fence();

  for (int g = blockIdx.x; g < N_NODES / 64; g += gridDim.x) {
    const int r0 = g * 64 + wv * 16;
    __syncthreads();

    // xs = agg_s @ w_down0
    for (int i = lane; i < 16 * (F / 8); i += 32) {
      int r = i >> 4, c8 = (i & 15) << 3;
      const float* src = agg_s + (size_t)(r0 + r) * F + c8;
      v8bf v;
#pragma unroll
      for (int j = 0; j < 8; ++j) v[j] = (__bf16)src[j];
      *(v8bf*)(s_in[wv] + r * F + c8) = v;
      *(v4f*)(s_vn[wv] + r * F + c8)     = (v4f){0.f, 0.f, 0.f, 0.f};
      *(v4f*)(s_vn[wv] + r * F + c8 + 4) = (v4f){0.f, 0.f, 0.f, 0.f};
    }
    __syncthreads();
#pragma unroll 1
    for (int t = 0; t < 8; ++t) {
      v8f acc = {};
#pragma unroll
      for (int ks = 0; ks < 4; ++ks)
        acc = wmma_bf16(load_a_frag(s_wd0 + t * 16 * F, F, ks * 32, lane),
                        load_b_act(s_in[wv], F, ks * 32, lane), acc);
      float* xp = s_xs[wv] + col * F + t * 16 + mg;
      v4f o0, o1;
#pragma unroll
      for (int r = 0; r < 4; ++r) { o0[r] = acc[r] * INV128; o1[r] = acc[4 + r] * INV128; }
      *(v4f*)xp = o0; *(v4f*)(xp + 4) = o1;
    }
    __syncthreads();

    // vnorm = sum_c (agg_v_c @ w_down1)^2
    for (int c = 0; c < 3; ++c) {
      for (int i = lane; i < 16 * (F / 8); i += 32) {
        int r = i >> 4, c8 = (i & 15) << 3;
        const float* src = agg_v + (size_t)c * NF + (size_t)(r0 + r) * F + c8;
        v8bf v;
#pragma unroll
        for (int j = 0; j < 8; ++j) v[j] = (__bf16)src[j];
        *(v8bf*)(s_in[wv] + r * F + c8) = v;
      }
      __syncthreads();
#pragma unroll 1
      for (int t = 0; t < 8; ++t) {
        v8f acc = {};
#pragma unroll
        for (int ks = 0; ks < 4; ++ks)
          acc = wmma_bf16(load_a_frag(s_wd1 + t * 16 * F, F, ks * 32, lane),
                          load_b_act(s_in[wv], F, ks * 32, lane), acc);
        float* vp = s_vn[wv] + col * F + t * 16 + mg;
#pragma unroll
        for (int r = 0; r < 8; ++r) { float v = acc[r] * INV128; vp[r] += v * v; }
      }
      __syncthreads();
    }

    // symmetric contraction (correlation<=2), species-indexed
    for (int i = lane; i < 16 * (F / 8); i += 32) {
      int r = i >> 4, c8 = (i & 15) << 3;
      int n = r0 + r;
      int z = species[n];
      const float* wz = w_sc + (size_t)z * 3 * F + c8;
      v8bf o;
#pragma unroll
      for (int j = 0; j < 8; ++j) {
        float xs = s_xs[wv][r * F + c8 + j];
        float fp = (wz[j] * xs + wz[F + j] * xs * xs + wz[2 * F + j] * s_vn[wv][r * F + c8 + j]) * INVZ;
        o[j] = (__bf16)fp;
      }
      *(v8bf*)(s_in[wv] + r * F + c8) = o;
    }
    __syncthreads();

    // feats = fp @ w_post * inv + self_conn
#pragma unroll 1
    for (int t = 0; t < 8; ++t) {
      v8f acc = {};
#pragma unroll
      for (int ks = 0; ks < 4; ++ks)
        acc = wmma_bf16(load_a_frag(s_wp + t * 16 * F, F, ks * 32, lane),
                        load_b_act(s_in[wv], F, ks * 32, lane), acc);
      const float* scp = self_conn + (size_t)(r0 + col) * F + t * 16 + mg;
      float* fop = feats_out + (size_t)(r0 + col) * F + t * 16 + mg;
      float* sxp = s_xs[wv] + col * F + t * 16 + mg;
#pragma unroll
      for (int r = 0; r < 8; ++r) {
        float v = acc[r] * INV128 + scp[r];
        fop[r] = v;
        sxp[r] = v;
      }
    }
    __syncthreads();

    // readout = silu(feats @ w_read1) @ w_read2
    for (int i = lane; i < 16 * (F / 8); i += 32) {
      int r = i >> 4, c8 = (i & 15) << 3;
      v8bf v;
#pragma unroll
      for (int j = 0; j < 8; ++j) v[j] = (__bf16)s_xs[wv][r * F + c8 + j];
      *(v8bf*)(s_in[wv] + r * F + c8) = v;
    }
    __syncthreads();
    {
      v8f acc = {};
#pragma unroll
      for (int ks = 0; ks < 4; ++ks)
        acc = wmma_bf16(load_a_frag(s_wr1, F, ks * 32, lane),
                        load_b_act(s_in[wv], F, ks * 32, lane), acc);
      float* hp = s_vn[wv] + col * 16 + mg;
#pragma unroll
      for (int r = 0; r < 8; ++r) hp[r] = silu_f(acc[r] * INV128);
    }
    __syncthreads();
    if (lane < 16) {
      float o = 0.f;
#pragma unroll
      for (int c2 = 0; c2 < RW; ++c2) o += s_vn[wv][lane * 16 + c2] * s_wr2[c2];
      out[r0 + lane] = o * INVR;
    }
  }
}

// ---------------------------------------------------------------------------
extern "C" void kernel_launch(void* const* d_in, const int* in_sizes, int n_in,
                              void* d_out, int out_size, void* d_ws, size_t ws_size,
                              hipStream_t stream) {
  (void)in_sizes; (void)n_in; (void)out_size; (void)ws_size;

  const float* vectors      = (const float*)d_in[0];
  const float* node_scalars = (const float*)d_in[1];
  const float* node_vectors = (const float*)d_in[2];
  const float* radial       = (const float*)d_in[3];
  const float* w_skip       = (const float*)d_in[4];
  const float* w_up0        = (const float*)d_in[5];
  const float* w_up1        = (const float*)d_in[6];
  const float* mlp_w1       = (const float*)d_in[7];
  const float* mlp_w2       = (const float*)d_in[8];
  const float* mlp_w3       = (const float*)d_in[9];
  const float* mlp_w4       = (const float*)d_in[10];
  const float* w_down0      = (const float*)d_in[11];
  const float* w_down1      = (const float*)d_in[12];
  const float* w_sc         = (const float*)d_in[13];
  const float* w_post       = (const float*)d_in[14];
  const float* w_read1      = (const float*)d_in[15];
  const float* w_read2      = (const float*)d_in[16];
  const int*   senders      = (const int*)d_in[17];
  const int*   receivers    = (const int*)d_in[18];
  const int*   species      = (const int*)d_in[19];

  float* ws        = (float*)d_ws;
  float* hs        = ws;                       // [N,F]
  float* hv        = ws + (size_t)NF;          // [3][N,F] planar
  float* self_conn = ws + (size_t)4 * NF;      // [N,F]
  float* agg_s     = ws + (size_t)5 * NF;      // [N,F]
  float* agg_v     = ws + (size_t)6 * NF;      // [3][N,F] planar
  __bf16* wb       = (__bf16*)(ws + (size_t)9 * NF);  // bf16 weight arena

  float* out   = (float*)d_out;                // [N]
  float* feats = out + N_NODES;                // [N,F]

  // zero atomic accumulators every call (graph-replay deterministic work)
  (void)hipMemsetAsync(agg_s, 0, (size_t)4 * NF * sizeof(float), stream);

  k_prep_weights<<<64, 256, 0, stream>>>(w_up0, w_up1, mlp_w1, mlp_w2, mlp_w3, mlp_w4,
                                         w_down0, w_down1, w_post, w_read1, wb);
  k_node_up<<<128, 256, 0, stream>>>(node_scalars, node_vectors, wb, hs, hv);
  k_self_conn<<<N_NODES / 2, 256, 0, stream>>>(node_scalars, w_skip, species, self_conn);
  k_edge<<<2048, 128, 0, stream>>>(vectors, radial, wb, senders, receivers, hs, hv, agg_s, agg_v);
  k_node_down<<<256, 128, 0, stream>>>(agg_s, agg_v, wb, w_sc, w_read2,
                                       self_conn, species, out, feats);
}
